// DRNetTest_70102456205407
// MI455X (gfx1250) — matrix operand order repaired and checked by reading, not compiled, not run on hardware
//
#include <hip/hip_runtime.h>

typedef __attribute__((ext_vector_type(16))) __bf16 v16bf;
typedef __attribute__((ext_vector_type(8)))  float  v8f;

constexpr int HD     = 256;   // hidden size H
constexpr int TSTEPS = 256;   // scan length T
constexpr int ROWS_PER_WAVE  = 16;
constexpr int WAVES_PER_BLK  = 2;   // 165KB LDS -> 1 block/WGP, 128 WGPs busy
constexpr int ROWS_PER_BLK   = ROWS_PER_WAVE * WAVES_PER_BLK;  // 32
constexpr int WPAD = 264;     // padded bf16 row stride for W_hh (bank-conflict break)
constexpr int LPAD = 264;     // padded f32 row stride for latent

__device__ __forceinline__ float fast_tanh(float x) {
#if __has_builtin(__builtin_amdgcn_tanhf)
  return __builtin_amdgcn_tanhf(x);   // native v_tanh_f32 (TRANS, co-executes with WMMA)
#else
  // tanh(x) = 1 - 2/(exp2(2*log2e*x) + 1)
  float e = __builtin_amdgcn_exp2f(2.8853900818f * x);
  return 1.0f - 2.0f * __builtin_amdgcn_rcpf(e + 1.0f);
#endif
}

__device__ __forceinline__ v16bf load_afrag(const float* __restrict__ arow, int k0, int half) {
  v16bf a;
  const int base = k0 + half * 16;
  #pragma unroll
  for (int e = 0; e < 8; ++e) {
    a[e]     = (__bf16)arow[base + e];
    a[e + 8] = (__bf16)arow[base + 8 + e];
  }
  return a;
}

__device__ __forceinline__ v16bf load_bfrag(const __bf16* __restrict__ Wb, int g, int k0, int half) {
  const __bf16* p = &Wb[g * WPAD + k0 + half * 16];
  v16bf b;
  #pragma unroll
  for (int e = 0; e < 16; ++e) b[e] = p[e];
  return b;
}

// One K-step (K=32) of the 16x256 x 256x256 tile GEMM with a distance-2
// software pipeline on the B fragments so LDS latency overlaps WMMA execution.
template <bool FIRST>
__device__ __forceinline__ void gemm_ks(const float* __restrict__ arow,
                                        const __bf16* __restrict__ Wb,
                                        int k0, int n, int half, v8f acc[16]) {
  v16bf afrag = load_afrag(arow, k0, half);
  v16bf bfr0 = load_bfrag(Wb, 0 * 16 + n, k0, half);
  v16bf bfr1 = load_bfrag(Wb, 1 * 16 + n, k0, half);
  #pragma unroll
  for (int nt = 0; nt < 16; ++nt) {
    v16bf bnext;
    if (nt + 2 < 16) bnext = load_bfrag(Wb, (nt + 2) * 16 + n, k0, half);  // load leads by 2
    v16bf bcur = (nt & 1) ? bfr1 : bfr0;
    if (FIRST) {
      v8f z = {};
      acc[nt] = __builtin_amdgcn_wmma_f32_16x16x32_bf16(
          false, afrag, false, bcur, (short)0, z, false, false);
    } else {
      acc[nt] = __builtin_amdgcn_wmma_f32_16x16x32_bf16(
          false, afrag, false, bcur, (short)0, acc[nt], false, false);
    }
    if (nt + 2 < 16) { if (nt & 1) bfr1 = bnext; else bfr0 = bnext; }
  }
}

__global__ void __launch_bounds__(WAVES_PER_BLK * 32, 1)
drnet_scan_kernel(const float* __restrict__ flat_img,
                  const float* __restrict__ W_ih,
                  const float* __restrict__ W_hh,
                  const float* __restrict__ b_ih,
                  const float* __restrict__ b_hh,
                  float* __restrict__ out)
{
  extern __shared__ __align__(16) char smem[];
  __bf16* Wb  = (__bf16*)smem;                                       // [HD][WPAD] bf16
  float*  lat = (float*)(smem + (size_t)HD * WPAD * sizeof(__bf16)); // [32][LPAD] f32

  const int tid  = threadIdx.x;
  const int lane = tid & 31;
  const int wave = tid >> 5;
  const int n    = lane & 15;   // column in N tile / row in A tile
  const int half = lane >> 4;   // wave32 half select
  const int rowbase  = wave * ROWS_PER_WAVE;
  const int growbase = blockIdx.x * ROWS_PER_BLK + rowbase;
  const unsigned growshift = (unsigned)growbase << 8;  // row id premixed for noise hash

  // Stage W_hh into LDS as bf16 with padded rows.
  for (int idx = tid; idx < HD * HD; idx += blockDim.x) {
    int g = idx >> 8, h = idx & (HD - 1);
    Wb[g * WPAD + h] = (__bf16)W_hh[idx];
  }
  // LDS latent = 0 (published copy; registers hold the working copy)
  for (int idx = tid; idx < ROWS_PER_BLK * LPAD; idx += blockDim.x)
    lat[idx] = 0.0f;
  __syncthreads();

  // Per-lane column params, fixed across all steps.
  float wih[16], bias[16];
  #pragma unroll
  for (int nt = 0; nt < 16; ++nt) {
    int g = nt * 16 + n;
    wih[nt]  = W_ih[g];
    bias[nt] = b_ih[g] + b_hh[g];
  }

  // Register-resident latent in C-fragment layout (128 VGPRs; wave32 MSB file).
  v8f latreg[16];
  #pragma unroll
  for (int nt = 0; nt < 16; ++nt) { v8f z = {}; latreg[nt] = z; }

  const float* arow = &lat[(rowbase + n) * LPAD];

  for (int t = 0; t < TSTEPS; ++t) {
    float ts  = (float)(TSTEPS - t);
    float at  = (ts + 1.0f) / (float)(TSTEPS + 1);
    float sig = __builtin_sqrtf(at * (1.0f - at));
    // fold 0.1*sqrt(3) (uniform->0.1*N(0,1) variance match) and 2^-31 int scaling
    float nscale = sig * 0.17320508f * (1.0f / 2147483648.0f);
    unsigned tkey = (unsigned)t * 0x9E3779B1u;

    // lane s holds x_t of global row growbase + (s&15)
    float xv = flat_img[(size_t)(growbase + n) * TSTEPS + t];
    // prefetch next step's x column (speculative; OOB prefetch is dropped by HW)
    __builtin_prefetch(&flat_img[(size_t)(growbase + n) * TSTEPS + t + 1], 0, 1);

    v8f acc[16];
    gemm_ks<true>(arow, Wb, 0, n, half, acc);          // first WMMA uses inline-0 C
    for (int ks = 1; ks < 8; ++ks)
      gemm_ks<false>(arow, Wb, ks * 32, n, half, acc);

    // Only 8 distinct broadcasts exist per wave half.
    float xr8[8];
    #pragma unroll
    for (int j = 0; j < 8; ++j) xr8[j] = __shfl(xv, j + half * 8, 32);

    // Update: pure register math on latreg; publish to LDS for next step's A transpose.
    #pragma unroll
    for (int nt = 0; nt < 16; ++nt) {
      const int g = nt * 16 + n;
      #pragma unroll
      for (int j = 0; j < 8; ++j) {
        const int m = j + half * 8;
        float xr  = xr8[j];
        float pre = acc[nt][j] + xr * wih[nt] + bias[nt];
        float hv  = fast_tanh(pre);
        unsigned u = (growshift + (unsigned)((m << 8) | g)) ^ tkey;
        u *= 0x85EBCA6Bu;
        u ^= u >> 16;
        u *= 0x7FEB352Du;
        float lnew = latreg[nt][j] + xr + hv + (float)(int)u * nscale;
        latreg[nt][j] = lnew;
        lat[(rowbase + m) * LPAD + g] = lnew;   // ds_store_b32 (no read-back needed)
      }
    }
    // No barrier: each wave owns its latent rows; Wb is read-only; per-wave LDS is in-order.
  }
  __syncthreads();

  // Emit latent (1, B, H) f32 from the LDS copy.
  for (int idx = tid; idx < ROWS_PER_BLK * HD; idx += blockDim.x) {
    int r = idx >> 8, c = idx & (HD - 1);
    out[(size_t)(blockIdx.x * ROWS_PER_BLK + r) * HD + c] = lat[r * LPAD + c];
  }
}

extern "C" void kernel_launch(void* const* d_in, const int* in_sizes, int n_in,
                              void* d_out, int out_size, void* d_ws, size_t ws_size,
                              hipStream_t stream) {
  // inputs: [0]=T (scalar), [1]=flat_img (B,T) f32, [2]=W_ih (H,1), [3]=W_hh (H,H),
  //         [4]=b_ih (H,), [5]=b_hh (H,)
  const float* flat_img = (const float*)d_in[1];
  const float* W_ih     = (const float*)d_in[2];
  const float* W_hh     = (const float*)d_in[3];
  const float* b_ih     = (const float*)d_in[4];
  const float* b_hh     = (const float*)d_in[5];
  float* out = (float*)d_out;

  const int B = in_sizes[1] / TSTEPS;  // 4096
  const size_t lds_bytes = (size_t)HD * WPAD * sizeof(__bf16)
                         + (size_t)ROWS_PER_BLK * LPAD * sizeof(float); // ~165 KB < 320 KB WGP LDS

  dim3 grid(B / ROWS_PER_BLK);          // 128 blocks, each scans 32 rows for all 256 steps
  dim3 block(WAVES_PER_BLK * 32);       // 2 waves/block
  drnet_scan_kernel<<<grid, block, lds_bytes, stream>>>(flat_img, W_ih, W_hh, b_ih, b_hh, out);
}
